// attention_unit_73735998538275
// MI455X (gfx1250) — matrix operand order repaired and checked by reading, not compiled, not run on hardware
//
#include <hip/hip_runtime.h>
#include <hip/hip_bf16.h>
#include <math.h>

// Problem constants
#define BB 8
#define CC 130
#define CP 144        // h channels padded to 9*16 for WMMA tiling
#define NN 4096
#define CQ 32
#define BN_EPS 1e-5f

typedef __attribute__((ext_vector_type(16))) __bf16 v16bf;
typedef __attribute__((ext_vector_type(8)))  __bf16 v8bf;
typedef __attribute__((ext_vector_type(8)))  float  v8f;

union BF16x16 { v16bf v; v8bf h[2]; };

static __device__ __forceinline__ v8f v8f_zero() {
  v8f z = {0.f,0.f,0.f,0.f,0.f,0.f,0.f,0.f};
  return z;
}

// ---------------------------------------------------------------------------
// Kernel 1: 1x1-conv projections + BN + ReLU.
//   f -> fT bf16 [B][N][32]   (transposed: row n, contiguous c)
//   g -> gT bf16 [B][N][32]
//   h -> hb bf16 [B][144][N]  (row-major, rows 130..143 zero)
// Block = 256 threads handles one (b, 64-wide n chunk); x tile staged in LDS.
// ---------------------------------------------------------------------------
__global__ __launch_bounds__(256) void proj_kernel(
    const float* __restrict__ x,
    const float* __restrict__ wF, const float* __restrict__ bF,
    const float* __restrict__ bnFw, const float* __restrict__ bnFb,
    const float* __restrict__ bnFm, const float* __restrict__ bnFv,
    const float* __restrict__ wG, const float* __restrict__ bG,
    const float* __restrict__ bnGw, const float* __restrict__ bnGb,
    const float* __restrict__ bnGm, const float* __restrict__ bnGv,
    const float* __restrict__ wH, const float* __restrict__ bH,
    const float* __restrict__ bnHw, const float* __restrict__ bnHb,
    const float* __restrict__ bnHm, const float* __restrict__ bnHv,
    __bf16* __restrict__ fT, __bf16* __restrict__ gT, __bf16* __restrict__ hb)
{
  __shared__ float xs[CC * 64];
  const int blk = blockIdx.x;
  const int b  = blk >> 6;
  const int n0 = (blk & 63) << 6;
  const float* xb = x + ((size_t)b * CC) * NN + n0;

  for (int i = threadIdx.x; i < CC * 64; i += 256) {
    int c = i >> 6, nn = i & 63;
    xs[i] = xb[(size_t)c * NN + nn];
  }
  __syncthreads();

  // 194 output channels: [0,32) -> F, [32,64) -> G, [64,194) -> H
  for (int t = threadIdx.x; t < 194 * 64; t += 256) {
    const int o = t >> 6, nn = t & 63;
    const float* wr;
    if (o < 32)       wr = wF + o * CC;
    else if (o < 64)  wr = wG + (o - 32) * CC;
    else              wr = wH + (o - 64) * CC;

    float dot = 0.f;
    for (int c = 0; c < CC; ++c) dot += wr[c] * xs[(c << 6) + nn];

    float y;
    if (o < 32) {
      const int q = o;
      float sc = bnFw[q] * rsqrtf(bnFv[q] + BN_EPS);
      y = fmaxf((dot + bF[q] - bnFm[q]) * sc + bnFb[q], 0.f);
      fT[((size_t)b * NN + n0 + nn) * CQ + q] = (__bf16)y;
    } else if (o < 64) {
      const int q = o - 32;
      float sc = bnGw[q] * rsqrtf(bnGv[q] + BN_EPS);
      y = fmaxf((dot + bG[q] - bnGm[q]) * sc + bnGb[q], 0.f);
      gT[((size_t)b * NN + n0 + nn) * CQ + q] = (__bf16)y;
    } else {
      const int q = o - 64;
      float sc = bnHw[q] * rsqrtf(bnHv[q] + BN_EPS);
      y = fmaxf((dot + bH[q] - bnHm[q]) * sc + bnHb[q], 0.f);
      hb[((size_t)b * CP + q) * NN + n0 + nn] = (__bf16)y;
    }
  }

  // zero-fill padded h rows 130..143
  for (int t = threadIdx.x; t < (CP - CC) * 64; t += 256) {
    int row = CC + (t >> 6);
    hb[((size_t)b * CP + row) * NN + n0 + (t & 63)] = (__bf16)0.f;
  }
}

// ---------------------------------------------------------------------------
// Kernel 2 (pass 1): per-row softmax stats over s[n, :] = (G^T F)[n, :].
// One wave per (b, 16-row n tile). A = G tile (fixed), loop m in 16-wide B
// tiles of F; per-lane online max/expsum (each lane sees cols m%16==lane%16),
// final log-sum-exp merge across the 16-lane group via shuffles.
// WMMA shape 16x16x32 bf16: K=32 == CQ, one instruction per S tile.
// ---------------------------------------------------------------------------
__global__ __launch_bounds__(32) void rowstats_kernel(
    const __bf16* __restrict__ gT, const __bf16* __restrict__ fT,
    float* __restrict__ Mrow, float* __restrict__ rZ)
{
  const int b    = blockIdx.x >> 8;
  const int n0   = (blockIdx.x & 255) << 4;
  const int lane = threadIdx.x;
  const int lm   = lane & 15;   // A row / D col position
  const int hh   = lane >> 4;   // lane half (K split: this code uses contiguous K 16*hh..16*hh+15)

  // A operand: 16x32 bf16. lanes 0-15: elems[0..15]=K0..15;
  // lanes 16-31: K16..31.  gT rows are contiguous in c(=K).
  const __bf16* ga = gT + ((size_t)b * NN + n0 + lm) * CQ;
  BF16x16 A;
  A.h[0] = *(const v8bf*)(ga + 16 * hh);
  A.h[1] = *(const v8bf*)(ga + 16 * hh + 8);

  float Mr[8], Zr[8];
#pragma unroll
  for (int r = 0; r < 8; ++r) { Mr[r] = -__builtin_inff(); Zr[r] = 0.f; }

  for (int m0 = 0; m0 < NN; m0 += 16) {
    // B operand: 32x16 bf16. lanes 0-15 hold K0..15, lanes 16-31 K16..31;
    // column n = lane%16.  fT rows contiguous in c(=K).
    const __bf16* fa = fT + ((size_t)b * NN + m0 + lm) * CQ + 16 * hh;
    BF16x16 Bm;
    Bm.h[0] = *(const v8bf*)(fa);
    Bm.h[1] = *(const v8bf*)(fa + 8);

    v8f s = __builtin_amdgcn_wmma_f32_16x16x32_bf16(
        false, A.v, false, Bm.v, (short)0, v8f_zero(), false, false);

#pragma unroll
    for (int r = 0; r < 8; ++r) {          // row n0 + r + 8*hh, col m0 + lm
      float v  = s[r];
      float nM = fmaxf(Mr[r], v);
      Zr[r] = Zr[r] * __expf(Mr[r] - nM) + __expf(v - nM);
      Mr[r] = nM;
    }
  }

  // merge (M,Z) across the 16-lane group
#pragma unroll
  for (int r = 0; r < 8; ++r) {
    float M = Mr[r], Z = Zr[r];
#pragma unroll
    for (int mask = 1; mask < 16; mask <<= 1) {
      float oM = __shfl_xor(M, mask, 32);
      float oZ = __shfl_xor(Z, mask, 32);
      float nM = fmaxf(M, oM);
      Z = Z * __expf(M - nM) + oZ * __expf(oM - nM);
      M = nM;
    }
    Mr[r] = M; Zr[r] = Z;
  }

  if (lm == 0) {
    const int base = b * NN + n0 + 8 * hh;   // lane0 -> rows 0..7, lane16 -> 8..15
#pragma unroll
    for (int r = 0; r < 8; ++r) {
      Mrow[base + r] = Mr[r];
      rZ[base + r]   = 1.f / Zr[r];
    }
  }
}

// ---------------------------------------------------------------------------
// Kernel 3 (pass 2): O[:, m-tile] = sum_n H[:, n] * exp(s[n,m]-M[n])*rZ[n].
// One wave per (b, 16-wide m tile). B operand of the S-matmul (F tile) is
// fixed; loop n in steps of 32: two WMMAs rebuild S tiles, exp via v_exp_f32,
// bf16 P staged through LDS to convert C/D layout -> B layout, then 9 WMMAs
// accumulate the 144(=130 padded)-channel output tile. Residual + gamma fused.
// LDS ordering: single wave + hardware-in-order DS ops, so no s_barrier is
// needed; sched_barrier(0) only pins compiler ordering around Pt.
// ---------------------------------------------------------------------------
__global__ __launch_bounds__(32) void attn_out_kernel(
    const float* __restrict__ x,
    const __bf16* __restrict__ gT, const __bf16* __restrict__ fT,
    const __bf16* __restrict__ hb,
    const float* __restrict__ Mrow, const float* __restrict__ rZ,
    const float* __restrict__ gamma, float* __restrict__ out)
{
  __shared__ __attribute__((aligned(32))) __bf16 Pt[16 * 32];  // [m col][k=n]
  const int b    = blockIdx.x >> 8;
  const int m0   = (blockIdx.x & 255) << 4;
  const int lane = threadIdx.x;
  const int lm   = lane & 15;
  const int hh   = lane >> 4;

  // fixed B operand: F columns m0..m0+15 (32x16, K=c)
  const __bf16* fa = fT + ((size_t)b * NN + m0 + lm) * CQ + 16 * hh;
  BF16x16 Bf;
  Bf.h[0] = *(const v8bf*)(fa);
  Bf.h[1] = *(const v8bf*)(fa + 8);

  v8f acc[9];
#pragma unroll
  for (int t = 0; t < 9; ++t) acc[t] = v8f_zero();

  const __bf16* gbase = gT + ((size_t)b * NN + lm) * CQ;

  for (int n0 = 0; n0 < NN; n0 += 32) {
    // prefetch next iteration's G rows (global_prefetch_b8; no LOADcnt cost)
    if (n0 + 32 < NN)
      __builtin_prefetch(gbase + (size_t)(n0 + 32) * CQ, 0, 0);

#pragma unroll
    for (int t = 0; t < 2; ++t) {
      const int nb = n0 + 16 * t;
      // A operand: G rows nb..nb+15 (16x32, K=c)
      const __bf16* ga = gbase + (size_t)nb * CQ;
      BF16x16 Ag;
      Ag.h[0] = *(const v8bf*)(ga + 16 * hh);
      Ag.h[1] = *(const v8bf*)(ga + 16 * hh + 8);

      v8f s = __builtin_amdgcn_wmma_f32_16x16x32_bf16(
          false, Ag.v, false, Bf.v, (short)0, v8f_zero(), false, false);

      const float4 Ma = *(const float4*)(Mrow + (size_t)b * NN + nb + 8 * hh);
      const float4 Mb = *(const float4*)(Mrow + (size_t)b * NN + nb + 8 * hh + 4);
      const float4 Za = *(const float4*)(rZ   + (size_t)b * NN + nb + 8 * hh);
      const float4 Zb = *(const float4*)(rZ   + (size_t)b * NN + nb + 8 * hh + 4);
      const float mr[8] = {Ma.x, Ma.y, Ma.z, Ma.w, Mb.x, Mb.y, Mb.z, Mb.w};
      const float zr[8] = {Za.x, Za.y, Za.z, Za.w, Zb.x, Zb.y, Zb.z, Zb.w};

#pragma unroll
      for (int r = 0; r < 8; ++r) {  // S row k = r+8*hh (local), col = lm
        float p = __expf(s[r] - mr[r]) * zr[r];
        Pt[lm * 32 + 16 * t + 8 * hh + r] = (__bf16)p;
      }
    }
    __builtin_amdgcn_sched_barrier(0);  // keep DS stores before DS loads

    // P as B operand (32x16, K=n local); DS in-order per wave makes the
    // cross-lane store->load exchange safe without s_barrier.
    BF16x16 Bp;
    Bp.h[0] = *(const v8bf*)(&Pt[lm * 32 + 16 * hh]);
    Bp.h[1] = *(const v8bf*)(&Pt[lm * 32 + 16 * hh + 8]);

#pragma unroll
    for (int ct = 0; ct < 9; ++ct) {
      // A operand: H rows ct*16..+15, K = n0..n0+31 (contiguous in memory)
      const __bf16* ha = hb + ((size_t)b * CP + ct * 16 + lm) * NN + n0;
      BF16x16 Ah;
      Ah.h[0] = *(const v8bf*)(ha + 16 * hh);
      Ah.h[1] = *(const v8bf*)(ha + 16 * hh + 8);
      acc[ct] = __builtin_amdgcn_wmma_f32_16x16x32_bf16(
          false, Ah.v, false, Bp.v, (short)0, acc[ct], false, false);
    }
    __builtin_amdgcn_sched_barrier(0);  // keep DS loads before next stores
  }

  const float ga = gamma[0];
#pragma unroll
  for (int ct = 0; ct < 9; ++ct) {
#pragma unroll
    for (int r = 0; r < 8; ++r) {
      const int c = ct * 16 + 8 * hh + r;
      if (c < CC) {
        const size_t idx = ((size_t)b * CC + c) * NN + m0 + lm;
        out[idx] = ga * acc[ct][r] + x[idx];
      }
    }
  }
}

// ---------------------------------------------------------------------------
// Launcher. Workspace layout (~13.4 MB, all power-of-two aligned):
//   fT bf16 [8][4096][32]  (2 MB)
//   gT bf16 [8][4096][32]  (2 MB)
//   hb bf16 [8][144][4096] (9 MB)
//   M  f32  [8][4096]      (128 KB)
//   rZ f32  [8][4096]      (128 KB)
// ---------------------------------------------------------------------------
extern "C" void kernel_launch(void* const* d_in, const int* in_sizes, int n_in,
                              void* d_out, int out_size, void* d_ws, size_t ws_size,
                              hipStream_t stream) {
  const float* x     = (const float*)d_in[0];
  const float* wF    = (const float*)d_in[1];
  const float* bF    = (const float*)d_in[2];
  const float* bnFw  = (const float*)d_in[3];
  const float* bnFb  = (const float*)d_in[4];
  const float* bnFm  = (const float*)d_in[5];
  const float* bnFv  = (const float*)d_in[6];
  const float* wG    = (const float*)d_in[7];
  const float* bG    = (const float*)d_in[8];
  const float* bnGw  = (const float*)d_in[9];
  const float* bnGb  = (const float*)d_in[10];
  const float* bnGm  = (const float*)d_in[11];
  const float* bnGv  = (const float*)d_in[12];
  const float* wH    = (const float*)d_in[13];
  const float* bH    = (const float*)d_in[14];
  const float* bnHw  = (const float*)d_in[15];
  const float* bnHb  = (const float*)d_in[16];
  const float* bnHm  = (const float*)d_in[17];
  const float* bnHv  = (const float*)d_in[18];
  const float* gamma = (const float*)d_in[19];
  float* out = (float*)d_out;

  char* ws = (char*)d_ws;
  __bf16* fT = (__bf16*)ws;
  __bf16* gT = fT + (size_t)BB * NN * CQ;
  __bf16* hb = gT + (size_t)BB * NN * CQ;
  float*  Mr = (float*)(hb + (size_t)BB * CP * NN);
  float*  rZ = Mr + (size_t)BB * NN;

  proj_kernel<<<BB * (NN / 64), 256, 0, stream>>>(
      x, wF, bF, bnFw, bnFb, bnFm, bnFv,
      wG, bG, bnGw, bnGb, bnGm, bnGv,
      wH, bH, bnHw, bnHb, bnHm, bnHv,
      fT, gT, hb);

  rowstats_kernel<<<BB * (NN / 16), 32, 0, stream>>>(gT, fT, Mr, rZ);

  attn_out_kernel<<<BB * (NN / 16), 32, 0, stream>>>(
      x, gT, fT, hb, Mr, rZ, gamma, out);
}
